// part_seg1_15264313770020
// MI455X (gfx1250) — matrix operand order repaired and checked by reading, not compiled, not run on hardware
//
#include <hip/hip_runtime.h>
#include <hip/hip_bf16.h>
#include <math.h>

typedef __attribute__((ext_vector_type(16))) _Float16 v16h;
typedef __attribute__((ext_vector_type(8)))  float    v8f;

#define N_NODES 8192
#define N_EDGES 131072

// ---------------------------------------------------------------------------
// Weight prep: W fp32 [K,M] row-major -> Wt f16 [Mpad,K] (transposed, zero pad)
// ---------------------------------------------------------------------------
__global__ void wprep_kernel(const float* __restrict__ W, _Float16* __restrict__ Wt,
                             int K, int M, int Mpad) {
    int idx = blockIdx.x * blockDim.x + threadIdx.x;
    if (idx >= K * Mpad) return;
    int m = idx / K;
    int k = idx - m * K;
    Wt[idx] = (m < M) ? (_Float16)W[(size_t)k * M + m] : (_Float16)0.0f;
}

__global__ void biaspad_kernel(const float* __restrict__ b, float* __restrict__ bp,
                               int M, int Mpad) {
    int i = blockIdx.x * blockDim.x + threadIdx.x;
    if (i < Mpad) bp[i] = (i < M) ? b[i] : 0.0f;
}

// ---------------------------------------------------------------------------
// Degree / dinv  (deg = #incoming edges + 1 self loop; dinv = 1/sqrt(deg))
// ---------------------------------------------------------------------------
__global__ void deg_init_kernel(float* __restrict__ deg, int n) {
    int i = blockIdx.x * blockDim.x + threadIdx.x;
    if (i < n) deg[i] = 1.0f;               // self loop
}
__global__ void deg_count_kernel(const int* __restrict__ dst, float* __restrict__ deg, int e) {
    int i = blockIdx.x * blockDim.x + threadIdx.x;
    if (i < e) atomicAdd(&deg[dst[i]], 1.0f);
}
__global__ void dinv_kernel(const float* __restrict__ deg, float* __restrict__ dinv, int n) {
    int i = blockIdx.x * blockDim.x + threadIdx.x;
    if (i < n) dinv[i] = 1.0f / sqrtf(deg[i]);
}

// ---------------------------------------------------------------------------
// cnn1: x fp32 [N,3] @ W [3,32] + b, relu -> h f16 [N,32]
// ---------------------------------------------------------------------------
__global__ void cnn1_kernel(const float* __restrict__ x, const float* __restrict__ W,
                            const float* __restrict__ b, _Float16* __restrict__ h, int n) {
    int idx = blockIdx.x * blockDim.x + threadIdx.x;
    if (idx >= n * 32) return;
    int node = idx >> 5;
    int m    = idx & 31;
    float s = b[m]
            + x[node * 3 + 0] * W[0 * 32 + m]
            + x[node * 3 + 1] * W[1 * 32 + m]
            + x[node * 3 + 2] * W[2 * 32 + m];
    h[(size_t)node * 32 + m] = (_Float16)fmaxf(s, 0.0f);
}

// ---------------------------------------------------------------------------
// WMMA GEMM: out[N,M] = X[N,K](f16) @ W[K,M] (W stored transposed f16 [M,K])
// One wave -> 32x64 tile: 2 A fragments x 4 B fragments -> 8 accumulators.
// Requirements: K multiple of 32, M multiple of 64, N multiple of 32.
// ---------------------------------------------------------------------------
template <bool HAS_BIAS, bool RELU, bool OUT_F32>
__global__ __launch_bounds__(256)
void gemm_wmma_kernel(const _Float16* __restrict__ X, const _Float16* __restrict__ Wt,
                      const float* __restrict__ bias, void* __restrict__ outp,
                      int K, int M, int rtiles, int mtiles) {
    int lane = threadIdx.x & 31;
    int wave = threadIdx.x >> 5;
    int tile = blockIdx.x * 8 + wave;
    if (tile >= rtiles * mtiles) return;          // wave-uniform exit
    int rt = tile % rtiles;                       // consecutive waves: same col tile
    int mt = tile / rtiles;
    int row0 = rt * 32;
    int col0 = mt * 64;

    int kA = (lane >> 4) * 8;                     // A k sub-offset: 0 or 8 (+16 hi)
    int kB = (lane >> 4) * 8;                     // B k sub-offset: 0 or 8 (+16 hi)

    const _Float16* xrow0 = X + (size_t)(row0 + (lane & 15)) * K;
    const _Float16* xrow1 = xrow0 + (size_t)16 * K;
    const _Float16* wr0   = Wt + (size_t)(col0 + (lane & 15)) * K + kB;

    v8f acc[8];
#pragma unroll
    for (int i = 0; i < 8; ++i) acc[i] = v8f{};

    for (int k0 = 0; k0 < K; k0 += 32) {
        union { v16h v; uint4 q[2]; } a0, a1;
        a0.q[0] = *(const uint4*)(xrow0 + k0 + kA);
        a0.q[1] = *(const uint4*)(xrow0 + k0 + kA + 16);
        a1.q[0] = *(const uint4*)(xrow1 + k0 + kA);
        a1.q[1] = *(const uint4*)(xrow1 + k0 + kA + 16);
#pragma unroll
        for (int j = 0; j < 4; ++j) {
            union { v16h v; uint4 q[2]; } b;
            const _Float16* wp = wr0 + (size_t)j * 16 * K + k0;
            b.q[0] = *(const uint4*)(wp);
            b.q[1] = *(const uint4*)(wp + 16);
            acc[j]     = __builtin_amdgcn_wmma_f32_16x16x32_f16(false, a0.v, false, b.v, (short)0, acc[j],     false, false);
            acc[j + 4] = __builtin_amdgcn_wmma_f32_16x16x32_f16(false, a1.v, false, b.v, (short)0, acc[j + 4], false, false);
        }
    }

    int rb = (lane >> 4) << 3;                    // row sub-offset 0 or 8
#pragma unroll
    for (int j = 0; j < 4; ++j) {
        int c = col0 + j * 16 + (lane & 15);
        float bv = HAS_BIAS ? bias[c] : 0.0f;
#pragma unroll
        for (int v = 0; v < 8; ++v) {
            float v0 = acc[j][v]     + bv;
            float v1 = acc[j + 4][v] + bv;
            if (RELU) { v0 = fmaxf(v0, 0.0f); v1 = fmaxf(v1, 0.0f); }
            size_t i0 = (size_t)(row0 + rb + v) * M + c;
            size_t i1 = (size_t)(row0 + 16 + rb + v) * M + c;
            if (OUT_F32) {
                ((float*)outp)[i0] = v0;
                ((float*)outp)[i1] = v1;
            } else {
                ((_Float16*)outp)[i0] = (_Float16)v0;
                ((_Float16*)outp)[i1] = (_Float16)v1;
            }
        }
    }
}

// ---------------------------------------------------------------------------
// Zero fp32 buffer
// ---------------------------------------------------------------------------
__global__ void zero_kernel(float4* __restrict__ p, int n4) {
    int i = blockIdx.x * blockDim.x + threadIdx.x;
    if (i < n4) p[i] = float4{0.f, 0.f, 0.f, 0.f};
}

// ---------------------------------------------------------------------------
// GCN scatter: msg e in [0,E) uses (src,dst); e in [E,E+N) is self-loop.
// Thread handles one message x 4 features. M in {256,512,1024}.
// ---------------------------------------------------------------------------
struct h4 { _Float16 a, b, c, d; };

__global__ __launch_bounds__(256)
void gcn_scatter_kernel(const _Float16* __restrict__ hl, const int* __restrict__ src,
                        const int* __restrict__ dst, const float* __restrict__ dinv,
                        float* __restrict__ acc, int E, int N, int M, int chunk_shift) {
    long long tid = (long long)blockIdx.x * blockDim.x + threadIdx.x;
    int msg = (int)(tid >> chunk_shift);
    int ch  = (int)(tid & ((1 << chunk_shift) - 1));
    if (msg >= E + N) return;
    int s, d;
    if (msg < E) { s = src[msg]; d = dst[msg]; }
    else         { s = msg - E;  d = s; }
    float nrm = dinv[s] * dinv[d];
    int f = ch << 2;
    h4 hv = *(const h4*)(hl + (size_t)s * M + f);
    float* o = acc + (size_t)d * M + f;
    atomicAdd(o + 0, (float)hv.a * nrm);
    atomicAdd(o + 1, (float)hv.b * nrm);
    atomicAdd(o + 2, (float)hv.c * nrm);
    atomicAdd(o + 3, (float)hv.d * nrm);
}

// ---------------------------------------------------------------------------
// bias + relu + convert fp32 acc -> f16 activations (M is power of two)
// ---------------------------------------------------------------------------
__global__ void bias_relu_kernel(const float* __restrict__ acc, const float* __restrict__ bias,
                                 _Float16* __restrict__ h, int total, int mmask) {
    int i = blockIdx.x * blockDim.x + threadIdx.x;
    if (i >= total) return;
    int m = i & mmask;
    h[i] = (_Float16)fmaxf(acc[i] + bias[m], 0.0f);
}

// ---------------------------------------------------------------------------
// Softmax over C=50 classes (logits stored with row stride `stride`)
// ---------------------------------------------------------------------------
__global__ __launch_bounds__(256)
void softmax_kernel(const float* __restrict__ logits, float* __restrict__ out,
                    int n, int C, int stride) {
    int wave = threadIdx.x >> 5;
    int lane = threadIdx.x & 31;
    int row = blockIdx.x * 8 + wave;
    if (row >= n) return;
    const float* L = logits + (size_t)row * stride;
    float v0 = (lane      < C) ? L[lane]      : -INFINITY;
    float v1 = (lane + 32 < C) ? L[lane + 32] : -INFINITY;
    float m = fmaxf(v0, v1);
    for (int off = 16; off; off >>= 1) m = fmaxf(m, __shfl_xor(m, off, 32));
    float e0 = (lane      < C) ? expf(v0 - m) : 0.0f;
    float e1 = (lane + 32 < C) ? expf(v1 - m) : 0.0f;
    float s = e0 + e1;
    for (int off = 16; off; off >>= 1) s += __shfl_xor(s, off, 32);
    float inv = 1.0f / s;
    if (lane      < C) out[(size_t)row * C + lane]      = e0 * inv;
    if (lane + 32 < C) out[(size_t)row * C + lane + 32] = e1 * inv;
}

// ---------------------------------------------------------------------------
// Host side
// ---------------------------------------------------------------------------
static void launch_gemm(const _Float16* X, const _Float16* Wt, const float* bias, void* out,
                        int K, int Mpad, int mode /*0: bias+relu f16, 1: plain f16, 2: bias f32*/,
                        hipStream_t s) {
    const int N = N_NODES;
    int rtiles = N / 32;
    int mtiles = Mpad / 64;
    int waves  = rtiles * mtiles;
    int blocks = (waves + 7) / 8;
    if (mode == 0)
        gemm_wmma_kernel<true, true, false><<<blocks, 256, 0, s>>>(X, Wt, bias, out, K, Mpad, rtiles, mtiles);
    else if (mode == 1)
        gemm_wmma_kernel<false, false, false><<<blocks, 256, 0, s>>>(X, Wt, bias, out, K, Mpad, rtiles, mtiles);
    else
        gemm_wmma_kernel<true, false, true><<<blocks, 256, 0, s>>>(X, Wt, bias, out, K, Mpad, rtiles, mtiles);
}

extern "C" void kernel_launch(void* const* d_in, const int* in_sizes, int n_in,
                              void* d_out, int out_size, void* d_ws, size_t ws_size,
                              hipStream_t stream) {
    (void)in_sizes; (void)n_in; (void)out_size; (void)ws_size;
    const int N = N_NODES, E = N_EDGES;

    const float* x   = (const float*)d_in[0];
    const int*   ei  = (const int*)d_in[1];
    const int*   src = ei;
    const int*   dst = ei + E;

    // layer dims (K=din, M=dout); index 0 = cnn1 (handled separately)
    const int dims[10][2] = {
        {3, 32}, {32, 64}, {64, 128},
        {128, 256}, {256, 512}, {512, 1024},
        {1024, 512}, {512, 256}, {256, 128}, {128, 50}
    };
    const float* Wf[10]; const float* Bf[10];
    for (int i = 0; i < 10; ++i) {
        Wf[i] = (const float*)d_in[2 + 2 * i];
        Bf[i] = (const float*)d_in[3 + 2 * i];
    }

    // workspace carve-up
    char* ws = (char*)d_ws;
    _Float16* hA   = (_Float16*)(ws);                                   // 16 MB
    _Float16* hB   = (_Float16*)(ws + (size_t)(16u << 20));             // 16 MB
    float*    acc  = (float*)(ws + (size_t)(32u << 20));                // 32 MB (also final logits)
    float*    deg  = (float*)(ws + (size_t)(64u << 20));                // 32 KB
    float*    dinv = (float*)(ws + (size_t)(64u << 20) + 65536);        // 32 KB
    float*    bpad = (float*)(ws + (size_t)(64u << 20) + 131072);       // 64 floats (padded final bias)
    _Float16* wcur = (_Float16*)(ws + (size_t)(64u << 20) + 131072 + 512); // ~2.8 MB f16 weights

    // 1) weight prep (transpose + f16); final layer padded 50 -> 64 cols
    _Float16* Wt[10];
    for (int i = 1; i < 10; ++i) {
        int K = dims[i][0], M = dims[i][1];
        int Mpad = (M + 63) / 64 * 64;
        Wt[i] = wcur;
        wcur += (size_t)K * Mpad;
        int elems = K * Mpad;
        wprep_kernel<<<(elems + 255) / 256, 256, 0, stream>>>(Wf[i], Wt[i], K, M, Mpad);
    }
    biaspad_kernel<<<1, 64, 0, stream>>>(Bf[9], bpad, 50, 64);

    // 2) degree + dinv
    deg_init_kernel<<<(N + 255) / 256, 256, 0, stream>>>(deg, N);
    deg_count_kernel<<<(E + 255) / 256, 256, 0, stream>>>(dst, deg, E);
    dinv_kernel<<<(N + 255) / 256, 256, 0, stream>>>(deg, dinv, N);

    // 3) cnn1: x -> hA [N,32]
    cnn1_kernel<<<(N * 32 + 255) / 256, 256, 0, stream>>>(x, Wf[0], Bf[0], hA, N);

    // 4) cnn2, cnn3 (bias+relu, f16 out)
    launch_gemm(hA, Wt[1], Bf[1], hB, 32, 64, 0, stream);     // hB [N,64]
    launch_gemm(hB, Wt[2], Bf[2], hA, 64, 128, 0, stream);    // hA [N,128]

    // 5) GCN layers: hl = h@W (f16), scatter-add with norm, +bias, relu
    const int gcn_layers[3] = {3, 4, 5};
    _Float16* hcur = hA;
    _Float16* hl   = hB;
    for (int li = 0; li < 3; ++li) {
        int i = gcn_layers[li];
        int K = dims[i][0], M = dims[i][1];
        launch_gemm(hcur, Wt[i], nullptr, hl, K, M, 1, stream);            // hl f16 [N,M]
        int n4 = N * M / 4;
        zero_kernel<<<(n4 + 255) / 256, 256, 0, stream>>>((float4*)acc, n4);
        int shift = (M == 256) ? 6 : (M == 512) ? 7 : 8;                   // log2(M/4)
        long long tthreads = (long long)(E + N) << shift;
        int blocks = (int)((tthreads + 255) / 256);
        gcn_scatter_kernel<<<blocks, 256, 0, stream>>>(hl, src, dst, dinv, acc, E, N, M, shift);
        bias_relu_kernel<<<(N * M + 255) / 256, 256, 0, stream>>>(acc, Bf[i], hcur, N * M, M - 1);
        // hcur holds activations [N,M]; hl is scratch again
    }

    // 6) decoder: cnnd1..cnnd3 (bias+relu f16), cnnd4 (bias, fp32 logits into acc, stride 64)
    launch_gemm(hcur, Wt[6], Bf[6], hl,   1024, 512, 0, stream);   // hl  [N,512]
    launch_gemm(hl,   Wt[7], Bf[7], hcur, 512,  256, 0, stream);   // hcur[N,256]
    launch_gemm(hcur, Wt[8], Bf[8], hl,   256,  128, 0, stream);   // hl  [N,128]
    launch_gemm(hl,   Wt[9], bpad,  acc,  128,  64,  2, stream);   // acc fp32 [N,64] (50 valid)

    // 7) softmax -> d_out
    softmax_kernel<<<(N + 7) / 8, 256, 0, stream>>>(acc, (float*)d_out, N, 50, 64);
}
